// QLinear_790273983138
// MI455X (gfx1250) — 8-bit matrix operand order repaired and checked by reading, compiled, not run on hardware
//
#include <hip/hip_runtime.h>
#include <stdint.h>

typedef __attribute__((ext_vector_type(8))) int v8i;

#define QMAX 127.0f
#define EPSQ 1e-12f

__global__ void init_amax_kernel(unsigned int* amax_bits) {
    amax_bits[0] = 0u;
    amax_bits[1] = 0u;
}

__global__ __launch_bounds__(256) void amax_kernel(const float* __restrict__ x,
                                                   long long n4,
                                                   unsigned int* __restrict__ out_bits) {
    __shared__ float red[256];
    const float4* x4 = (const float4*)x;
    float m = 0.0f;
    long long i = (long long)blockIdx.x * blockDim.x + threadIdx.x;
    long long stride = (long long)gridDim.x * blockDim.x;
    for (long long j = i; j < n4; j += stride) {
        float4 v = x4[j];
        m = fmaxf(m, fmaxf(fmaxf(fabsf(v.x), fabsf(v.y)),
                           fmaxf(fabsf(v.z), fabsf(v.w))));
    }
    red[threadIdx.x] = m;
    __syncthreads();
    for (int s = 128; s > 0; s >>= 1) {
        if ((int)threadIdx.x < s)
            red[threadIdx.x] = fmaxf(red[threadIdx.x], red[threadIdx.x + s]);
        __syncthreads();
    }
    if (threadIdx.x == 0) {
        atomicMax(out_bits, __float_as_uint(red[0]));
    }
}

__global__ __launch_bounds__(256) void quant_kernel(const float* __restrict__ x,
                                                    long long n4,
                                                    const float* __restrict__ amax,
                                                    int8_t* __restrict__ q) {
    const float scale = fmaxf(*amax, EPSQ) / QMAX;
    const float inv = 1.0f / scale;
    const float4* x4 = (const float4*)x;
    unsigned int* q4 = (unsigned int*)q;
    long long i = (long long)blockIdx.x * blockDim.x + threadIdx.x;
    long long stride = (long long)gridDim.x * blockDim.x;
    for (long long j = i; j < n4; j += stride) {
        float4 v = x4[j];
        int q0 = (int)fminf(fmaxf(rintf(v.x * inv), -QMAX), QMAX);
        int q1 = (int)fminf(fmaxf(rintf(v.y * inv), -QMAX), QMAX);
        int q2 = (int)fminf(fmaxf(rintf(v.z * inv), -QMAX), QMAX);
        int q3 = (int)fminf(fmaxf(rintf(v.w * inv), -QMAX), QMAX);
        unsigned int pack = (unsigned int)(q0 & 0xff) |
                            ((unsigned int)(q1 & 0xff) << 8) |
                            ((unsigned int)(q2 & 0xff) << 16) |
                            ((unsigned int)(q3 & 0xff) << 24);
        q4[j] = pack;
    }
}

__global__ __launch_bounds__(256) void gemm_iu8_kernel(const int8_t* __restrict__ qx,
                                                       const int8_t* __restrict__ qw,
                                                       const float* __restrict__ amax,
                                                       const float* __restrict__ bias,
                                                       float* __restrict__ out,
                                                       int M, int N, int K) {
    const int lane  = threadIdx.x & 31;
    const int wave  = threadIdx.x >> 5;
    const int waveM = wave & 3;
    const int waveN = wave >> 2;
    const int l16   = lane & 15;
    const int hi    = lane >> 4;

    const int mWave = blockIdx.y * 128 + waveM * 32;
    const int nWave = blockIdx.x * 128 + waveN * 64;

    const size_t Ks = (size_t)K;
    const int8_t* aBase0 = qx + (size_t)(mWave + l16) * Ks + (hi ? 16 : 0);
    const int8_t* aBase1 = aBase0 + 16 * Ks;
    const int8_t* bBase0 = qw + (size_t)(nWave + l16) * Ks + (hi ? 16 : 0);
    const int8_t* bBase1 = bBase0 + 16 * Ks;
    const int8_t* bBase2 = bBase0 + 32 * Ks;
    const int8_t* bBase3 = bBase0 + 48 * Ks;

    v8i acc[2][4] = {};

    for (int kb = 0; kb < K; kb += 64) {
        v8i a0, a1, b0, b1, b2, b3;
        {
            const int* p;
            p = (const int*)(aBase0 + kb);      a0[0] = p[0]; a0[1] = p[1];
            p = (const int*)(aBase0 + kb + 8); a0[2] = p[0]; a0[3] = p[1];
            p = (const int*)(aBase0 + kb + 32); a0[4] = p[0]; a0[5] = p[1];
            p = (const int*)(aBase0 + kb + 40); a0[6] = p[0]; a0[7] = p[1];
            p = (const int*)(aBase1 + kb);      a1[0] = p[0]; a1[1] = p[1];
            p = (const int*)(aBase1 + kb + 8); a1[2] = p[0]; a1[3] = p[1];
            p = (const int*)(aBase1 + kb + 32); a1[4] = p[0]; a1[5] = p[1];
            p = (const int*)(aBase1 + kb + 40); a1[6] = p[0]; a1[7] = p[1];
        }
        {
            int4 t;
            t = *(const int4*)(bBase0 + kb);      b0[0] = t.x; b0[1] = t.y; b0[2] = t.z; b0[3] = t.w;
            t = *(const int4*)(bBase0 + kb + 32); b0[4] = t.x; b0[5] = t.y; b0[6] = t.z; b0[7] = t.w;
            t = *(const int4*)(bBase1 + kb);      b1[0] = t.x; b1[1] = t.y; b1[2] = t.z; b1[3] = t.w;
            t = *(const int4*)(bBase1 + kb + 32); b1[4] = t.x; b1[5] = t.y; b1[6] = t.z; b1[7] = t.w;
            t = *(const int4*)(bBase2 + kb);      b2[0] = t.x; b2[1] = t.y; b2[2] = t.z; b2[3] = t.w;
            t = *(const int4*)(bBase2 + kb + 32); b2[4] = t.x; b2[5] = t.y; b2[6] = t.z; b2[7] = t.w;
            t = *(const int4*)(bBase3 + kb);      b3[0] = t.x; b3[1] = t.y; b3[2] = t.z; b3[3] = t.w;
            t = *(const int4*)(bBase3 + kb + 32); b3[4] = t.x; b3[5] = t.y; b3[6] = t.z; b3[7] = t.w;
        }
        acc[0][0] = __builtin_amdgcn_wmma_i32_16x16x64_iu8(true, a0, true, b0, acc[0][0], false, false);
        acc[0][1] = __builtin_amdgcn_wmma_i32_16x16x64_iu8(true, a0, true, b1, acc[0][1], false, false);
        acc[0][2] = __builtin_amdgcn_wmma_i32_16x16x64_iu8(true, a0, true, b2, acc[0][2], false, false);
        acc[0][3] = __builtin_amdgcn_wmma_i32_16x16x64_iu8(true, a0, true, b3, acc[0][3], false, false);
        acc[1][0] = __builtin_amdgcn_wmma_i32_16x16x64_iu8(true, a1, true, b0, acc[1][0], false, false);
        acc[1][1] = __builtin_amdgcn_wmma_i32_16x16x64_iu8(true, a1, true, b1, acc[1][1], false, false);
        acc[1][2] = __builtin_amdgcn_wmma_i32_16x16x64_iu8(true, a1, true, b2, acc[1][2], false, false);
        acc[1][3] = __builtin_amdgcn_wmma_i32_16x16x64_iu8(true, a1, true, b3, acc[1][3], false, false);
    }

    const float s = (fmaxf(amax[0], EPSQ) / QMAX) * (fmaxf(amax[1], EPSQ) / QMAX);

#pragma unroll
    for (int mi = 0; mi < 2; ++mi) {
#pragma unroll
        for (int ni = 0; ni < 4; ++ni) {
            const int n = nWave + ni * 16 + l16;
            const float bv = bias[n];
            const int mBase = mWave + mi * 16 + hi * 8;
#pragma unroll
            for (int r = 0; r < 8; ++r) {
                out[(size_t)(mBase + r) * (size_t)N + n] = s * (float)acc[mi][ni][r] + bv;
            }
        }
    }
}

extern "C" void kernel_launch(void* const* d_in, const int* in_sizes, int n_in,
                              void* d_out, int out_size, void* d_ws, size_t ws_size,
                              hipStream_t stream) {
    const float* x    = (const float*)d_in[0];
    const float* w    = (const float*)d_in[1];
    const float* bias = (const float*)d_in[2];
    float* out = (float*)d_out;

    const long long xsz = in_sizes[0];
    const long long wsz = in_sizes[1];
    const int dOut = in_sizes[2];
    const int dIn  = (int)(wsz / dOut);
    const int tokens = (int)(xsz / dIn);

    int8_t* qx = (int8_t*)d_ws;
    int8_t* qw = qx + xsz;
    float* amax = (float*)(qw + wsz);

    init_amax_kernel<<<1, 1, 0, stream>>>((unsigned int*)amax);

    amax_kernel<<<2048, 256, 0, stream>>>(x, xsz / 4, (unsigned int*)(amax + 0));
    amax_kernel<<<2048, 256, 0, stream>>>(w, wsz / 4, (unsigned int*)(amax + 1));

    quant_kernel<<<4096, 256, 0, stream>>>(x, xsz / 4, amax + 0, qx);
    quant_kernel<<<4096, 256, 0, stream>>>(w, wsz / 4, amax + 1, qw);

    dim3 grid(dOut / 128, tokens / 128);
    gemm_iu8_kernel<<<grid, 256, 0, stream>>>(qx, qw, amax, bias, out,
                                              tokens, dOut, dIn);
}
